// MambaPlusPlus_layer_48103633715538
// MI455X (gfx1250) — matrix operand order repaired and checked by reading, not compiled, not run on hardware
//
#include <hip/hip_runtime.h>

// ---------------------------------------------------------------------------
// Types for CDNA5 WMMA (wave32)
// ---------------------------------------------------------------------------
typedef __attribute__((ext_vector_type(16))) __bf16 v16bf;
typedef __attribute__((ext_vector_type(8)))  float  v8f;
typedef unsigned short u16;
typedef int v4i_t __attribute__((vector_size(16)));   // matches builtin param type

union FragAB { v16bf v; uint4 q[2]; };   // 32 bytes = 16 bf16 = 8 VGPRs
union FragC  { v8f   v; float f[8]; };   // 8 VGPRs fp32 accumulator

__device__ __forceinline__ float bf2f(u16 h) {
    return __uint_as_float(((unsigned)h) << 16);
}
__device__ __forceinline__ u16 f2bf(float f) {
    unsigned u = __float_as_uint(f);
    unsigned r = u + 0x7FFFu + ((u >> 16) & 1u);   // round-to-nearest-even
    return (u16)(r >> 16);
}

// CDNA5 async global->LDS path (ASYNCcnt-tracked), with sync fallback.
#if defined(__gfx1250__) && __has_builtin(__builtin_amdgcn_global_load_async_to_lds_b128)
#define HAVE_ASYNC_LDS 1
#else
#define HAVE_ASYNC_LDS 0
#endif

__device__ __forceinline__ void async_fill_b128(const u16* g, u16* l) {
#if HAVE_ASYNC_LDS
    __builtin_amdgcn_global_load_async_to_lds_b128(
        (__attribute__((address_space(1))) v4i_t*)(g),
        (__attribute__((address_space(3))) v4i_t*)(l), 0, 0);
#else
    *reinterpret_cast<uint4*>(l) = *reinterpret_cast<const uint4*>(g);
#endif
}

__device__ __forceinline__ void wait_async_fill() {
#if HAVE_ASYNC_LDS
#if __has_builtin(__builtin_amdgcn_s_wait_asynccnt)
    __builtin_amdgcn_s_wait_asynccnt(0);
#else
    asm volatile("s_wait_asynccnt 0x0" ::: "memory");
#endif
#endif
}

// ---------------------------------------------------------------------------
// fp32 -> bf16 conversion
// ---------------------------------------------------------------------------
__global__ void cvt_bf16_kernel(const float* __restrict__ s, u16* __restrict__ d, int n) {
    int i = blockIdx.x * 256 + threadIdx.x;
    if (i < n) d[i] = f2bf(s[i]);
}

// ---------------------------------------------------------------------------
// WMMA GEMM:  out[M,N] = epi( A[M,K] * W[N,K]^T + bias[N] )
//   A, W bf16 row-major (lda/ldw = K stride).  Optional z-batch strides.
//   EPI: 0 = fp32 out, 1 = tanh -> bf16, 2 = identity -> bf16, 3 = gelu -> bf16
//
// Block: 256 threads = 8 waves, arranged 4 (M) x 2 (N).
// Block tile 128 x 128; wave tile 32 x 64 (2 A-frags x 4 B-frags, 8 accums).
// B tile (128 cols x 32 k) staged in LDS, double-buffered, filled with the
// async global->LDS path; A streams straight from global (per-wave unique).
// Requires M%128==0, N%128==0, K%32==0 (true for all launches here).
// ---------------------------------------------------------------------------
#define BPAD 36   // LDS row stride in u16 (72B) -> conflict-free ds_load_b128

template <int EPI>
__global__ __launch_bounds__(256) void wmma_gemm_kernel(
    const u16* __restrict__ A, const u16* __restrict__ W,
    const float* __restrict__ bias,
    float* __restrict__ Cf, u16* __restrict__ Cb,
    int M, int N, int K, int lda, int ldw, int ldc,
    long long sA, long long sW, long long sBias, long long sC)
{
    __shared__ u16 btile[2][128 * BPAD];

    const int zb = blockIdx.z;
    A    += (size_t)zb * sA;
    W    += (size_t)zb * sW;
    bias += (size_t)zb * sBias;

    const int wave = threadIdx.x >> 5;
    const int lane = threadIdx.x & 31;
    const int wm   = wave & 3;          // wave row (M)
    const int wn   = wave >> 2;         // wave col (N)
    const int row0 = blockIdx.y * 128 + wm * 32;
    const int colB = blockIdx.x * 128;  // block N offset
    const int colw = colB + wn * 64;    // wave N offset
    const int r    = lane & 15;
    const int ksel = lane >> 4;

    // cooperative B-tile fill mapping: 128 rows x 32 k, 2 threads per row
    const int fRow = threadIdx.x >> 1;        // 0..127
    const int fSeg = (threadIdx.x & 1) * 16;  // element offset 0 or 16
    const u16* fSrc = W + (size_t)(colB + fRow) * ldw + fSeg;
    u16* fDst0 = &btile[0][fRow * BPAD + fSeg];
    u16* fDst1 = &btile[1][fRow * BPAD + fSeg];

    FragC acc[2][4];
#pragma unroll
    for (int i = 0; i < 2; ++i)
#pragma unroll
        for (int j = 0; j < 4; ++j)
#pragma unroll
            for (int e = 0; e < 8; ++e) acc[i][j].f[e] = 0.f;

    // A 16x32 bf16 layout: lanes 0-15 hold K = [k0+ksel*8 .. +7] and [+16 ..]
    const u16* aRow0 = A + (size_t)(row0 + r) * lda + ksel * 8;
    const u16* aRow1 = aRow0 + (size_t)16 * lda;

    // prologue: fill buffer 0 for k0 = 0
    async_fill_b128(fSrc, fDst0);
    async_fill_b128(fSrc + 8, fDst0 + 8);

    int pb = 0;
    for (int k0 = 0; k0 < K; k0 += 32) {
        wait_async_fill();
        __syncthreads();                       // B tile (pb) visible to all waves

        if (k0 + 32 < K) {                     // overlap next fill with WMMA
            const u16* s = fSrc + k0 + 32;
            u16* d = pb ? fDst0 : fDst1;
            async_fill_b128(s, d);
            async_fill_b128(s + 8, d + 8);
        }

        FragAB af[2];
        {
            const u16* pa0 = aRow0 + k0;
            const u16* pa1 = aRow1 + k0;
            af[0].q[0] = *reinterpret_cast<const uint4*>(pa0);
            af[0].q[1] = *reinterpret_cast<const uint4*>(pa0 + 16);
            af[1].q[0] = *reinterpret_cast<const uint4*>(pa1);
            af[1].q[1] = *reinterpret_cast<const uint4*>(pa1 + 16);
            __builtin_prefetch(pa0 + 128, 0, 0);   // global_prefetch_b8
            __builtin_prefetch(pa1 + 128, 0, 0);
        }

#pragma unroll
        for (int j = 0; j < 4; ++j) {
            // B 32x16 frag: lane n holds col n, K = [ksel*8 .. +7] and [16+ksel*8 .. +7] (ds_load_b128)
            FragAB bfr;
            const u16* bp = &btile[pb][(wn * 64 + j * 16 + r) * BPAD + ksel * 8];
            bfr.q[0] = *reinterpret_cast<const uint4*>(bp);
            bfr.q[1] = *reinterpret_cast<const uint4*>(bp + 16);
#pragma unroll
            for (int i = 0; i < 2; ++i) {
                acc[i][j].v = __builtin_amdgcn_wmma_f32_16x16x32_bf16(
                    false, af[i].v, false, bfr.v, (short)0, acc[i][j].v, false, false);
            }
        }

        __syncthreads();                       // reads done before overwrite
        pb ^= 1;
    }

    // C/D layout: lane l -> col = (l&15); VGPR v -> row = v + 8*(l>>4)
#pragma unroll
    for (int i = 0; i < 2; ++i) {
        const int rbase = row0 + i * 16 + ksel * 8;
#pragma unroll
        for (int j = 0; j < 4; ++j) {
            const int c  = colw + j * 16 + r;
            const float bv = bias[c];
#pragma unroll
            for (int v = 0; v < 8; ++v) {
                float x = acc[i][j].f[v] + bv;
                if (EPI == 1) x = tanhf(x);
                else if (EPI == 3) x = 0.5f * x * (1.f + erff(x * 0.70710678118654752f));
                const size_t idx = (size_t)(rbase + v) * ldc + c;
                if (EPI == 0) {
                    float* out = Cf + (size_t)zb * sC;
                    out[idx] = x;
                } else {
                    u16* out = Cb + (size_t)zb * sC;
                    out[idx] = f2bf(x);
                }
            }
        }
    }
}

// ---------------------------------------------------------------------------
// Sequential recurrence h_t = a_t * h_{t-1} + b_t over L, parallel over
// B*H*D = 16384 channels.  Layout [B, L, H*D]: at each step the 8192
// adjacent-channel threads read contiguous memory (coalesced).
// ---------------------------------------------------------------------------
__global__ __launch_bounds__(256) void scan_kernel(
    const u16* __restrict__ aG, const u16* __restrict__ bG,
    u16* __restrict__ hs, int L)
{
    const int t  = blockIdx.x * 256 + threadIdx.x;  // 0 .. 16383
    const int bi = t >> 13;                         // / 8192
    const int c  = t & 8191;
    size_t idx = (size_t)bi * L * 8192 + c;
    float h = 0.f;
    for (int l = 0; l < L; ++l) {
        h = bf2f(aG[idx]) * h + bf2f(bG[idx]);
        hs[idx] = f2bf(h);
        idx += 8192;
    }
}

// ---------------------------------------------------------------------------
// Block (256-thread) sum reduction helper
// ---------------------------------------------------------------------------
__device__ __forceinline__ float blk_sum(float v) {
    __shared__ float sm[256];
    const int t = threadIdx.x;
    sm[t] = v;
    __syncthreads();
    for (int s = 128; s > 0; s >>= 1) {
        if (t < s) sm[t] += sm[t + s];
        __syncthreads();
    }
    float r = sm[0];
    __syncthreads();
    return r;
}

// ---------------------------------------------------------------------------
// Fused: out = (C-GEMM result) * Dx ; per-head LayerNorm(hn_g, hn_b);
// sum_heads = (Dx + sum_h head_w[h]*LN) / 8  -> bf16.  One block per token.
// ---------------------------------------------------------------------------
__global__ __launch_bounds__(256) void post_heads_kernel(
    const float* __restrict__ outF, const float* __restrict__ Dx,
    const float* __restrict__ hn_g, const float* __restrict__ hn_b,
    const float* __restrict__ head_w, u16* __restrict__ sumB)
{
    const int m = blockIdx.x;
    const int t = threadIdx.x;
    float dx[4], acc[4];
#pragma unroll
    for (int i = 0; i < 4; ++i) {
        dx[i]  = Dx[(size_t)m * 1024 + t + i * 256];
        acc[i] = 0.f;
    }
    for (int h = 0; h < 8; ++h) {
        const float* row = outF + (size_t)m * 8192 + h * 1024;
        float v[4], ps = 0.f;
#pragma unroll
        for (int i = 0; i < 4; ++i) {
            float x = row[t + i * 256] * dx[i];
            v[i] = x; ps += x;
        }
        const float mean = blk_sum(ps) * (1.f / 1024.f);
        float ps2 = 0.f;
#pragma unroll
        for (int i = 0; i < 4; ++i) { float d = v[i] - mean; ps2 += d * d; }
        const float var = blk_sum(ps2) * (1.f / 1024.f);
        const float rs  = rsqrtf(var + 1e-5f);
        const float hw  = head_w[h];
#pragma unroll
        for (int i = 0; i < 4; ++i) {
            const int c = h * 1024 + t + i * 256;
            acc[i] += hw * ((v[i] - mean) * rs * hn_g[c] + hn_b[c]);
        }
    }
#pragma unroll
    for (int i = 0; i < 4; ++i)
        sumB[(size_t)m * 1024 + t + i * 256] = f2bf((dx[i] + acc[i]) * 0.125f);
}

// ---------------------------------------------------------------------------
// u = z + LayerNorm(z, g, b); write fp32 (for residual) and bf16 (GEMM input)
// ---------------------------------------------------------------------------
__global__ __launch_bounds__(256) void ln_residual_kernel(
    const float* __restrict__ z, const float* __restrict__ g,
    const float* __restrict__ b, float* __restrict__ uF, u16* __restrict__ uB)
{
    const int m = blockIdx.x;
    const int t = threadIdx.x;
    float v[4], ps = 0.f;
#pragma unroll
    for (int i = 0; i < 4; ++i) {
        v[i] = z[(size_t)m * 1024 + t + i * 256];
        ps += v[i];
    }
    const float mean = blk_sum(ps) * (1.f / 1024.f);
    float ps2 = 0.f;
#pragma unroll
    for (int i = 0; i < 4; ++i) { float d = v[i] - mean; ps2 += d * d; }
    const float var = blk_sum(ps2) * (1.f / 1024.f);
    const float rs  = rsqrtf(var + 1e-5f);
#pragma unroll
    for (int i = 0; i < 4; ++i) {
        const int c = t + i * 256;
        const float u = v[i] + (v[i] - mean) * rs * g[c] + b[c];
        uF[(size_t)m * 1024 + c] = u;
        uB[(size_t)m * 1024 + c] = f2bf(u);
    }
}

// ---------------------------------------------------------------------------
// elementwise: o = bf16(a + b)
// ---------------------------------------------------------------------------
__global__ void add_bf16_kernel(const float* __restrict__ a,
                                const float* __restrict__ b,
                                u16* __restrict__ o, int n) {
    int i = blockIdx.x * 256 + threadIdx.x;
    if (i < n) o[i] = f2bf(a[i] + b[i]);
}

// ---------------------------------------------------------------------------
// Host launcher
// ---------------------------------------------------------------------------
extern "C" void kernel_launch(void* const* d_in, const int* in_sizes, int n_in,
                              void* d_out, int out_size, void* d_ws, size_t ws_size,
                              hipStream_t stream) {
    (void)in_sizes; (void)n_in; (void)out_size; (void)ws_size;
    const float* emb    = (const float*)d_in[0];
    const float* Wa_w   = (const float*)d_in[1];
    const float* Wa_b   = (const float*)d_in[2];
    const float* Wb_w   = (const float*)d_in[3];
    const float* Wb_b   = (const float*)d_in[4];
    const float* C_w    = (const float*)d_in[5];
    const float* C_b    = (const float*)d_in[6];
    const float* D_w    = (const float*)d_in[7];
    const float* D_b    = (const float*)d_in[8];
    const float* Wout_w = (const float*)d_in[9];
    const float* Wout_b = (const float*)d_in[10];
    const float* hn_g   = (const float*)d_in[11];
    const float* hn_b   = (const float*)d_in[12];
    const float* head_w = (const float*)d_in[13];
    const float* ffn1_w = (const float*)d_in[14];
    const float* ffn1_b = (const float*)d_in[15];
    const float* ffn2_w = (const float*)d_in[16];
    const float* ffn2_b = (const float*)d_in[17];
    const float* proj_w = (const float*)d_in[18];
    const float* proj_b = (const float*)d_in[19];
    const float* norm_g = (const float*)d_in[20];
    const float* norm_b = (const float*)d_in[21];
    float* out = (float*)d_out;

    const int M = 4096;           // B*L
    const int E = 1024, HD = 8192, F = 4096;

    // workspace bump allocator (256B aligned; base from hipMalloc is aligned)
    char* ws = (char*)d_ws;
    size_t off = 0;
    auto bump = [&](size_t bytes) -> char* {
        char* p = ws + off;
        off = (off + bytes + 255) & ~(size_t)255;
        return p;
    };
    u16*   embB  = (u16*)  bump((size_t)M * E * 2);
    u16*   WaB   = (u16*)  bump((size_t)HD * E * 2);
    u16*   WbB   = (u16*)  bump((size_t)HD * E * 2);
    u16*   CwB   = (u16*)  bump((size_t)HD * E * 2);
    u16*   DwB   = (u16*)  bump((size_t)E * E * 2);
    u16*   WoutB = (u16*)  bump((size_t)E * E * 2);
    u16*   f1B   = (u16*)  bump((size_t)F * E * 2);
    u16*   f2B   = (u16*)  bump((size_t)E * F * 2);
    u16*   prB   = (u16*)  bump((size_t)E * E * 2);
    u16*   aG    = (u16*)  bump((size_t)M * HD * 2);
    u16*   bG    = (u16*)  bump((size_t)M * HD * 2);
    u16*   hsB   = (u16*)  bump((size_t)M * HD * 2);
    float* DxF   = (float*)bump((size_t)M * E * 4);
    float* outF  = (float*)bump((size_t)M * HD * 4);
    u16*   sumB  = (u16*)  bump((size_t)M * E * 2);
    float* zF    = (float*)bump((size_t)M * E * 4);
    float* uF    = (float*)bump((size_t)M * E * 4);
    u16*   uB    = (u16*)  bump((size_t)M * E * 2);
    u16*   ffnB  = (u16*)  bump((size_t)M * F * 2);
    float* t2F   = (float*)bump((size_t)M * E * 4);

    const dim3 blk(256);
    auto cvt = [&](const float* s, u16* d, int n) {
        cvt_bf16_kernel<<<dim3((n + 255) / 256), blk, 0, stream>>>(s, d, n);
    };
    cvt(emb,    embB,  M * E);
    cvt(Wa_w,   WaB,   HD * E);
    cvt(Wb_w,   WbB,   HD * E);
    cvt(C_w,    CwB,   HD * E);
    cvt(D_w,    DwB,   E * E);
    cvt(Wout_w, WoutB, E * E);
    cvt(ffn1_w, f1B,   F * E);
    cvt(ffn2_w, f2B,   E * F);
    cvt(proj_w, prB,   E * E);

    // a = tanh(emb @ Wa^T + Wa_b)  -> bf16 [M, HD]
    wmma_gemm_kernel<1><<<dim3(HD / 128, M / 128, 1), blk, 0, stream>>>(
        embB, WaB, Wa_b, nullptr, aG, M, HD, E, E, E, HD, 0, 0, 0, 0);
    // b = emb @ Wb^T + Wb_b        -> bf16 [M, HD]
    wmma_gemm_kernel<2><<<dim3(HD / 128, M / 128, 1), blk, 0, stream>>>(
        embB, WbB, Wb_b, nullptr, bG, M, HD, E, E, E, HD, 0, 0, 0, 0);
    // Dx = emb @ D_w^T + D_b       -> fp32 [M, E]
    wmma_gemm_kernel<0><<<dim3(E / 128, M / 128, 1), blk, 0, stream>>>(
        embB, DwB, D_b, DxF, nullptr, M, E, E, E, E, E, 0, 0, 0, 0);

    // sequential scan over L (16384 independent channels)
    scan_kernel<<<dim3(64), blk, 0, stream>>>(aG, bG, hsB, 2048);

    // per-head: out[:,h,:] = hs[:,h,:] @ C_w[h]^T + C_b[h]  (batched over z=h)
    wmma_gemm_kernel<0><<<dim3(E / 128, M / 128, 8), blk, 0, stream>>>(
        hsB, CwB, C_b, outF, nullptr, M, E, E, HD, E, HD,
        (long long)E, (long long)E * E, (long long)E, (long long)E);

    // * Dx, per-head LN, head-weighted sum -> sum_heads bf16
    post_heads_kernel<<<dim3(M), blk, 0, stream>>>(outF, DxF, hn_g, hn_b, head_w, sumB);

    // z = sum_heads @ Wout^T + Wout_b -> fp32
    wmma_gemm_kernel<0><<<dim3(E / 128, M / 128, 1), blk, 0, stream>>>(
        sumB, WoutB, Wout_b, zF, nullptr, M, E, E, E, E, E, 0, 0, 0, 0);

    // u = z + LN(z)
    ln_residual_kernel<<<dim3(M), blk, 0, stream>>>(zF, norm_g, norm_b, uF, uB);

    // ffn = gelu(u @ ffn1^T + ffn1_b) -> bf16 [M, F]
    wmma_gemm_kernel<3><<<dim3(F / 128, M / 128, 1), blk, 0, stream>>>(
        uB, f1B, ffn1_b, nullptr, ffnB, M, F, E, E, E, F, 0, 0, 0, 0);

    // t2 = ffn @ ffn2^T + ffn2_b -> fp32 [M, E]
    wmma_gemm_kernel<0><<<dim3(E / 128, M / 128, 1), blk, 0, stream>>>(
        ffnB, f2B, ffn2_b, t2F, nullptr, M, E, F, F, F, E, 0, 0, 0, 0);

    // h_out = u + t2 -> bf16
    add_bf16_kernel<<<dim3((M * E + 255) / 256), blk, 0, stream>>>(uF, t2F, uB, M * E);

    // final: h_out @ proj^T + proj_b -> fp32 d_out
    wmma_gemm_kernel<0><<<dim3(E / 128, M / 128, 1), blk, 0, stream>>>(
        uB, prB, proj_b, out, nullptr, M, E, E, E, E, E, 0, 0, 0, 0);
}
